// CondConv2d_25606595018937
// MI455X (gfx1250) — matrix operand order repaired and checked by reading, not compiled, not run on hardware
//
#include <hip/hip_runtime.h>
#include <stdint.h>

typedef __attribute__((ext_vector_type(16))) __bf16 v16bf;
typedef __attribute__((ext_vector_type(8)))  float  v8f;
typedef unsigned int v4u __attribute__((ext_vector_type(4)));
typedef int          v8i __attribute__((ext_vector_type(8)));
typedef int          v4i __attribute__((ext_vector_type(4)));

#define BB 32
#define EE 8
#define OO 192
#define II 192
#define HH 56
#define WWD 56
#define KKK 1728   // I*3*3 (GEMM K), tap-major: k' = tap*192 + ci
#define NN 3136    // H*W   (GEMM N)
#define TM 64
#define TN 128
#define TK 32
#define LDP 8      // LDS row padding (bf16): row pitch 80B (= TDM pad 4 DWORDs)

#ifndef __has_builtin
#define __has_builtin(x) 0
#endif
#if __has_builtin(__builtin_amdgcn_tensor_load_to_lds) && \
    __has_builtin(__builtin_amdgcn_s_wait_tensorcnt)
#define HAVE_TDM 1
#else
#define HAVE_TDM 0
#endif

__device__ __forceinline__ uint16_t f32_to_bf16(float f) {
  uint32_t u = __builtin_bit_cast(uint32_t, f);
  u += 0x7FFFu + ((u >> 16) & 1u);        // round-to-nearest-even
  return (uint16_t)(u >> 16);
}

struct Frag32B { uint4 lo, hi; };  // 32 bytes = 16 bf16

// ---------------------------------------------------------------------------
// Kernel 1: mix expert weights -> bf16 [B][m][k'] with k' = tap*192 + ci.
// Each thread owns one (m,k'-pair), reads the 8 expert values once, then
// loops over all 32 samples: memory-optimal (weights read exactly once).
// ---------------------------------------------------------------------------
__global__ __launch_bounds__(256) void mix_weights_kernel(
    const float* __restrict__ rw,       // [B, E]
    const float* __restrict__ weight,   // [E, O*KKK] (k = ci*9 + tap order)
    uint32_t* __restrict__ wmixPairs)   // [B, O*KKK/2] packed bf16 pairs
{
  const int pid   = blockIdx.x * blockDim.x + threadIdx.x;  // 0..165887
  const int flat2 = pid * 2;                                // m*1728 + k'
  const int m     = flat2 / KKK;
  const int kp    = flat2 - m * KKK;                        // k' (even)
  const int tap   = kp / II;
  const int ci    = kp - tap * II;

  const size_t rowStride = (size_t)OO * KKK;
  const float* w0 = weight + (size_t)m * KKK + (size_t)ci * 9 + tap;

  float we0[EE], we1[EE];
  #pragma unroll
  for (int e = 0; e < EE; ++e) {
    we0[e] = w0[e * rowStride];        // ci
    we1[e] = w0[e * rowStride + 9];    // ci+1 (next k' element)
  }

  for (int b = 0; b < BB; ++b) {
    const float* rwb = rw + b * EE;
    float s0 = 0.f, s1 = 0.f;
    #pragma unroll
    for (int e = 0; e < EE; ++e) {
      float r = rwb[e];
      s0 += r * we0[e];
      s1 += r * we1[e];
    }
    uint32_t pk = (uint32_t)f32_to_bf16(s0) | ((uint32_t)f32_to_bf16(s1) << 16);
    wmixPairs[(size_t)b * (rowStride / 2) + pid] = pk;
  }
}

// ---------------------------------------------------------------------------
// Kernel 2: per-sample implicit-GEMM conv, v_wmma_f32_16x16x32_bf16.
// C[64x128] tile per workgroup; 8 wave32, each a 16x64 strip (4 WMMA/K-step).
// A tile staged by the Tensor Data Mover (wave 0), B tile by tap-major
// coalesced x gathers (tap math hoisted out of the K loop).
// ---------------------------------------------------------------------------
__global__ __launch_bounds__(256) void condconv_wmma_kernel(
    const float*    __restrict__ x,     // [B, I, H, W]
    const float*    __restrict__ rw,    // [B, E]
    const float*    __restrict__ biasw, // [E, O]
    const uint16_t* __restrict__ wmix,  // [B, O*KKK] bf16, tap-major k'
    float*          __restrict__ out)   // [B, O, H, W]
{
  __shared__ __align__(16) uint16_t Alds[TM][TK + LDP];   // 64 x 40 (A: [m][k])
  __shared__ __align__(16) uint16_t Blds[TN][TK + LDP];   // 128 x 40 (B^T: [n][k])

  const int b     = blockIdx.z;
  const int mBase = blockIdx.y * TM;        // 0,64,128
  const int nBase = blockIdx.x * TN;        // 0..3072 (tail guarded)
  const int t     = threadIdx.x;
  const int lane  = t & 31;
  const int wid   = t >> 5;                 // 0..7
  const int waveM = wid & 3;                // 16-row strip
  const int waveN = wid >> 2;               // 64-col strip
  const int lm    = lane & 15;
  const int lhalf = lane >> 4;

  // B staging: thread owns column n, 8 ci-pairs (half selects pair block)
  const int sn  = t & 127;
  const int sph = t >> 7;                   // 0/1
  const int nG  = nBase + sn;
  const int sh  = nG / WWD;
  const int sw  = nG - sh * WWD;

  // A staging (fallback path): one 16B chunk per thread
  const int ar = t >> 2;                    // row 0..63
  const int ac = (t & 3) * 8;               // col chunk 0,8,16,24
  const uint16_t* wmixA = wmix + (size_t)b * ((size_t)OO * KKK)
                               + (size_t)mBase * KKK;      // [mBase][0]
  const float* xb = x + (size_t)b * II * HH * WWD;

#if HAVE_TDM
  const uint32_t ldsA   = (uint32_t)(size_t)(&Alds[0][0]); // LDS byte offset
  const uint64_t gbaseA = (uint64_t)(size_t)wmixA;         // global byte base
#endif

  v8f acc[4] = {v8f{}, v8f{}, v8f{}, v8f{}};

  for (int tap = 0; tap < 9; ++tap) {
    const int dy = tap / 3 - 1;
    const int dx = tap - (tap / 3) * 3 - 1;
    const int hy = sh + dy;
    const int wx = sw + dx;
    const bool valid = ((unsigned)hy < (unsigned)HH) &
                       ((unsigned)wx < (unsigned)WWD);
    const float* xtap = xb + hy * WWD + wx;  // + ci*H*W per channel

    for (int kc = 0; kc < II / TK; ++kc) {   // 6 ci-blocks of 32
      const int k0 = tap * II + kc * TK;     // column offset in A row

      // ---- stage A tile: Tensor Data Mover (wave 0) or b128 fallback ----
#if HAVE_TDM
      if (wid == 0) {
        const uint64_t ga = gbaseA + (uint64_t)k0 * 2u;    // tile start (bytes)
        v4u g0;
        g0.x = 1u;                                         // count=1, user mode
        g0.y = ldsA;                                       // lds_addr
        g0.z = (uint32_t)ga;                               // global_addr lo
        g0.w = (uint32_t)((ga >> 32) & 0x01FFFFFFull)      // global_addr[56:32]
             | (2u << 30);                                 // type=2 ("image")
        v8i g1;
        g1[0] = (int)((1u << 16)        // data_size: 2 bytes
                    | (1u << 20)        // pad_enable
                    | (3u << 22)        // pad_interval: 16 DWORDs (64B rows)
                    | (3u << 25));      // pad_amount: 4 DWORDs (16B pad)
        g1[1] = (int)(((uint32_t)KKK & 0xFFFFu) << 16);    // tensor_dim0 lo16
        g1[2] = (int)((((uint32_t)KKK >> 16) & 0xFFFFu)    // tensor_dim0 hi16
                    | (((uint32_t)OO & 0xFFFFu) << 16));   // tensor_dim1 lo16
        g1[3] = (int)((32u << 16));                        // tile_dim0 = 32
        g1[4] = (int)(64u);                                // tile_dim1 = 64
        g1[5] = (int)KKK;                                  // dim0_stride lo32
        g1[6] = 0;
        g1[7] = 0;
        v4i gz = {0, 0, 0, 0};
#if __clang_major__ >= 23
        v8i gz8 = {0, 0, 0, 0, 0, 0, 0, 0};
        __builtin_amdgcn_tensor_load_to_lds(g0, g1, gz, gz, gz8, 0);
#else
        __builtin_amdgcn_tensor_load_to_lds(g0, g1, gz, gz, 0);
#endif
      }
#else
      {
        const uint16_t* src = wmixA + (size_t)ar * KKK + k0 + ac;
        uint4 aval = *(const uint4*)src;
        *(uint4*)(&Alds[ar][ac]) = aval;
        if (k0 + TK < KKK)
          __builtin_prefetch(src + TK, 0, 3);              // global_prefetch_b8
      }
#endif

      // ---- stage B tile: coalesced stride-HW x loads, bf16 pack ----
      const int ciBase = kc * TK;
      #pragma unroll
      for (int j = 0; j < 8; ++j) {
        const int p   = sph * 8 + j;             // pair index 0..15
        const int ci0 = ciBase + 2 * p;
        uint32_t pk = 0;
        if (valid) {
          float v0 = xtap[(size_t)ci0 * (HH * WWD)];
          float v1 = xtap[(size_t)(ci0 + 1) * (HH * WWD)];
          pk = (uint32_t)f32_to_bf16(v0) | ((uint32_t)f32_to_bf16(v1) << 16);
        }
        *(uint32_t*)(&Blds[sn][2 * p]) = pk;
      }

#if HAVE_TDM
      __builtin_amdgcn_s_wait_tensorcnt(0);      // A tile landed in LDS
#endif
      __syncthreads();

      // ---- A fragment (ISA 16-bit A 16x32 layout): two ds_load_b128 ----
      Frag32B afr;
      afr.lo = *(const uint4*)(&Alds[waveM * 16 + lm][lhalf * 8]);
      afr.hi = *(const uint4*)(&Alds[waveM * 16 + lm][16 + lhalf * 8]);
      v16bf a = __builtin_bit_cast(v16bf, afr);

      // ---- 4 WMMAs along N ----
      #pragma unroll
      for (int nb = 0; nb < 4; ++nb) {
        const int col = waveN * 64 + nb * 16 + lm;
        Frag32B bfr;
        bfr.lo = *(const uint4*)(&Blds[col][lhalf * 8]);
        bfr.hi = *(const uint4*)(&Blds[col][16 + lhalf * 8]);
        v16bf bfrag = __builtin_bit_cast(v16bf, bfr);
        acc[nb] = __builtin_amdgcn_wmma_f32_16x16x32_bf16(
            false, a, false, bfrag, (short)0, acc[nb], false, false);
      }
      __syncthreads();
    }
  }

  // ---- epilogue: add on-the-fly mixed bias, store f32 --------------------
  const float* rwb = rw + b * EE;
  float r0 = rwb[0], r1 = rwb[1], r2 = rwb[2], r3 = rwb[3];
  float r4 = rwb[4], r5 = rwb[5], r6 = rwb[6], r7 = rwb[7];

  #pragma unroll
  for (int r = 0; r < 8; ++r) {
    const int m = mBase + waveM * 16 + lhalf * 8 + r;   // C layout: M=r+8*half
    float bias = r0 * biasw[0 * OO + m] + r1 * biasw[1 * OO + m]
               + r2 * biasw[2 * OO + m] + r3 * biasw[3 * OO + m]
               + r4 * biasw[4 * OO + m] + r5 * biasw[5 * OO + m]
               + r6 * biasw[6 * OO + m] + r7 * biasw[7 * OO + m];
    #pragma unroll
    for (int nb = 0; nb < 4; ++nb) {
      const int col = nBase + waveN * 64 + nb * 16 + lm; // C layout: N=lane%16
      if (col < NN)
        out[((size_t)b * OO + m) * NN + col] = acc[nb][r] + bias;
    }
  }
}

// ---------------------------------------------------------------------------
extern "C" void kernel_launch(void* const* d_in, const int* in_sizes, int n_in,
                              void* d_out, int out_size, void* d_ws, size_t ws_size,
                              hipStream_t stream) {
  const float* x      = (const float*)d_in[0];  // [32,192,56,56]
  const float* rw     = (const float*)d_in[1];  // [32,8]
  const float* weight = (const float*)d_in[2];  // [8, 192*192*9]
  const float* biasw  = (const float*)d_in[3];  // [8,192]
  float* out = (float*)d_out;                   // [32,192,56,56]

  uint16_t* wmix = (uint16_t*)d_ws;             // [32, 331776] bf16 = 21.2 MB

  // one thread per (m, k'-pair): 331776/2 / 256 = 648 blocks
  mix_weights_kernel<<<dim3(648), 256, 0, stream>>>(
      rw, weight, (uint32_t*)wmix);

  // N tiles: ceil(3136/128)=25 ; M tiles: 192/64=3 ; batch 32
  condconv_wmma_kernel<<<dim3(25, 3, BB), 256, 0, stream>>>(
      x, rw, biasw, wmix, out);
}
